// GAGNN_V2_dipol_53034256171642
// MI455X (gfx1250) — matrix operand order repaired and checked by reading, not compiled, not run on hardware
//
#include <hip/hip_runtime.h>
#include <hip/hip_bf16.h>
#include <math.h>

#define NN 10000
#define EE 100000
#define GG 500
#define DD 128
#define RR 4
#define TT 5
#define KK 20
#define D5 640
#define RCUT 5.0f
#define PI_F 3.14159265358979f

typedef __bf16 bf16t;
typedef __attribute__((ext_vector_type(16))) __bf16 v16bf;
typedef __attribute__((ext_vector_type(8)))  float  v8f;

// ---------------------------------------------------------------- WMMA core
__device__ __forceinline__ v8f wmma_bf16(v16bf a, v16bf b, v8f c) {
  // D = A(16x32 bf16) * B(32x16 bf16) + C(16x16 f32)
  return __builtin_amdgcn_wmma_f32_16x16x32_bf16(
      /*neg_a=*/false, a, /*neg_b=*/false, b,
      /*c_mod=*/(short)0, c, /*reuse_a=*/false, /*reuse_b=*/false);
}

// A fragment (16x32, 16-bit) from a row-major bf16 tile (LDS).
// lane<16: M=lane,    K in {k0+0..7, k0+16..23}
// lane>=16:M=lane-16, K in {k0+8..15, k0+24..31}
__device__ __forceinline__ v16bf load_a_frag(const bf16t* A, int lda, int m0,
                                             int k0, int lane) {
  const bf16t* p = A + (m0 + (lane & 15)) * lda + k0 + ((lane >> 4) << 3);
  v16bf a;
#pragma unroll
  for (int h = 0; h < 8; ++h) { a[h] = p[h]; a[h + 8] = p[h + 16]; }
  return a;
}

// B fragment from the pre-packed buffer: 512 contiguous halves per fragment,
// lane's 16 halves contiguous -> one coalesced 32B load per lane.
__device__ __forceinline__ v16bf load_b_frag(const bf16t* Bp, int frag, int lane) {
  return *(const v16bf*)(Bp + ((size_t)frag << 9) + ((size_t)lane << 4));
}

__device__ __forceinline__ void prefetch_b(const bf16t* Bp, int frag, int lane) {
  __builtin_prefetch(Bp + ((size_t)frag << 9) + ((size_t)lane << 4), 0, 0);
}

// fast silu: x * v_rcp_f32(1 + e^-x)  (avoids IEEE div_scale/fixup sequence)
__device__ __forceinline__ float silu(float x) {
  return x * __builtin_amdgcn_rcpf(1.0f + __expf(-x));
}

// ------------------------------------------------- pack W (n_out x n_in) for
// out = act @ W^T.  B element = W[n][k];  n = nt*16+(lane&15),
// k = kt*32 + 8*(lane>>4) + (h&7) + 16*(h>>3).  frag index = nt*KT + kt (k-inner contiguous).
__global__ void k_pack_b(const float* __restrict__ W, bf16t* __restrict__ Bp,
                         int n_out, int n_in) {
  size_t idx = (size_t)blockIdx.x * blockDim.x + threadIdx.x;
  int KT = n_in >> 5;
  size_t total = (size_t)(n_out >> 4) * KT * 512;
  if (idx >= total) return;
  int h = (int)(idx & 15);
  int lane = (int)((idx >> 4) & 31);
  size_t frag = idx >> 9;
  int kt = (int)(frag % KT);
  int nt = (int)(frag / KT);
  int n = (nt << 4) + (lane & 15);
  int k = (kt << 5) + ((lane >> 4) << 3) + (h & 7) + ((h >> 3) << 4);
  Bp[idx] = (bf16t)W[(size_t)n * n_in + k];
}

// ---------------------------------------------------------------- utilities
__global__ void k_zero(float* __restrict__ p, size_t n) {
  size_t i = (size_t)blockIdx.x * blockDim.x + threadIdx.x;
  size_t st = (size_t)gridDim.x * blockDim.x;
  for (; i < n; i += st) p[i] = 0.f;
}

__global__ void k_add(float* __restrict__ a, const float* __restrict__ b, size_t n) {
  size_t i = (size_t)blockIdx.x * blockDim.x + threadIdx.x;
  size_t st = (size_t)gridDim.x * blockDim.x;
  for (; i < n; i += st) a[i] += b[i];
}

__global__ void k_init_state(const int* __restrict__ Z,
                             const float* __restrict__ emb_s,
                             const float* __restrict__ emb_t,
                             float* __restrict__ state) {
  int idx = blockIdx.x * blockDim.x + threadIdx.x;
  if (idx >= NN * DD) return;
  int n = idx >> 7, d = idx & 127;
  int z = Z[n];
  float* s = state + (size_t)n * 8 * DD;
  s[d] = emb_s[z * DD + d];
#pragma unroll
  for (int c = 1; c < 7; ++c) s[c * DD + d] = 0.f;
  s[7 * DD + d] = emb_t[z * DD + d];
}

// -------------------------------------------------------- fused edge kernel
// 32 edges/block, 8 waves.  GEMM1: H = silu(ss0 @ W1^T + b1) -> LDS bf16.
// GEMM2: phi = H @ W2^T + b2; gate with rbf_out*cutoff; scatter messages.
__global__ __launch_bounds__(256) void k_edge(
    const float* __restrict__ state, float* __restrict__ agg,
    const int* __restrict__ edge_list, const float* __restrict__ edge_lengths,
    const float* __restrict__ edge_vectors,
    const bf16t* __restrict__ W1p, const float* __restrict__ b1,
    const bf16t* __restrict__ W2p, const float* __restrict__ b2,
    const float* __restrict__ rbfW, const float* __restrict__ rbfb) {
  __shared__ bf16t sA[32 * 128];     //  8 KB
  __shared__ bf16t sH[32 * 640];     // 40 KB
  __shared__ float sRbf[32 * 20];    //  2.5 KB
  __shared__ float sCut[32];
  __shared__ int sSend[32], sRecv[32];

  int tid = threadIdx.x, lane = tid & 31, wave = tid >> 5;
  int e0 = blockIdx.x * 32;

  if (tid < 32) {
    int e = e0 + tid;
    sSend[tid] = edge_list[e * 2 + 0];
    sRecv[tid] = edge_list[e * 2 + 1];
    float r = fmaxf(edge_lengths[e], 1e-6f);
    sCut[tid] = 0.5f * (__cosf(PI_F * r / RCUT) + 1.0f) * (r < RCUT ? 1.0f : 0.0f);
  }
  for (int i = tid; i < 32 * 20; i += 256) {
    int m = i / 20, k = i % 20;
    float r = fmaxf(edge_lengths[e0 + m], 1e-6f);
    sRbf[i] = __sinf(r * (float)(k + 1) * (PI_F / RCUT)) * __builtin_amdgcn_rcpf(r);
  }
  __syncthreads();
  for (int i = tid; i < 32 * 128; i += 256) {
    int m = i >> 7, d = i & 127;
    sA[i] = (bf16t)state[(size_t)sSend[m] * 1024 + d];
  }
  __syncthreads();

  // GEMM1: 2 m-tiles x 40 n-tiles, K=128
  for (int f = wave; f < 80; f += 8) {
    int mt = f / 40, nt = f % 40;
    v8f acc = {};
#pragma unroll
    for (int kt = 0; kt < 4; ++kt) {
      v16bf a = load_a_frag(sA, 128, mt * 16, kt * 32, lane);
      v16bf b = load_b_frag(W1p, nt * 4 + kt, lane);
      acc = wmma_bf16(a, b, acc);
    }
    int n = nt * 16 + (lane & 15);
    int mb = mt * 16 + ((lane >> 4) << 3);
    float bias = b1[n];
#pragma unroll
    for (int j = 0; j < 8; ++j)
      sH[(mb + j) * 640 + n] = (bf16t)silu(acc[j] + bias);
  }
  __syncthreads();

  // GEMM2 (K=640) + gating + message scatter
  for (int f = wave; f < 80; f += 8) {
    int mt = f / 40, nt = f % 40;
    v8f acc = {};
    for (int kt = 0; kt < 20; ++kt) {
      if (kt + 1 < 20) prefetch_b(W2p, nt * 20 + kt + 1, lane);
      v16bf a = load_a_frag(sH, 640, mt * 16, kt * 32, lane);
      v16bf b = load_b_frag(W2p, nt * 20 + kt, lane);
      acc = wmma_bf16(a, b, acc);
    }
    int n = nt * 16 + (lane & 15);
    int mb = mt * 16 + ((lane >> 4) << 3);
    float bias = b2[n], rb = rbfb[n];
    int c = n >> 7, dd = n & 127;
#pragma unroll
    for (int j = 0; j < 8; ++j) {
      int m = mb + j;
      float ro = rb;
#pragma unroll
      for (int q = 0; q < 20; ++q) ro += sRbf[m * 20 + q] * rbfW[n * 20 + q];
      ro *= sCut[m];
      float g = (acc[j] + bias) * ro;
      int e = e0 + m;
      float* ar = agg + (size_t)sRecv[m] * 1024;
      const float* ss = state + (size_t)sSend[m] * 1024;
      if (c == 0) {
        atomicAdd(ar + dd, g);
      } else if (c == 1) {  // g_v * vector channels of sender
        atomicAdd(ar + 128 + dd, g * ss[128 + dd]);
        atomicAdd(ar + 256 + dd, g * ss[256 + dd]);
        atomicAdd(ar + 384 + dd, g * ss[384 + dd]);
      } else if (c == 2) {  // g_d * edge vector
        atomicAdd(ar + 128 + dd, g * edge_vectors[e * 3 + 0]);
        atomicAdd(ar + 256 + dd, g * edge_vectors[e * 3 + 1]);
        atomicAdd(ar + 384 + dd, g * edge_vectors[e * 3 + 2]);
      } else if (c == 3) {  // g_b * bivector channels
        atomicAdd(ar + 512 + dd, g * ss[512 + dd]);
        atomicAdd(ar + 640 + dd, g * ss[640 + dd]);
        atomicAdd(ar + 768 + dd, g * ss[768 + dd]);
      } else {              // g_t * trivector channel
        atomicAdd(ar + 896 + dd, g * ss[896 + dd]);
      }
    }
  }
}

// ------------------------------------------- U,V: (N*8 x 128) @ W^T, dual out
__global__ __launch_bounds__(256) void k_uv(
    const float* __restrict__ state, const bf16t* __restrict__ Up,
    const bf16t* __restrict__ Vp, float* __restrict__ U, float* __restrict__ V) {
  __shared__ bf16t sA[64 * 128];
  int tid = threadIdx.x, lane = tid & 31, wave = tid >> 5;
  size_t m0 = (size_t)blockIdx.x * 64;
  for (int i = tid; i < 64 * 128; i += 256)
    sA[i] = (bf16t)state[m0 * 128 + i];
  __syncthreads();
  for (int f = wave; f < 64; f += 8) {
    int uv = f >> 5;
    int mt = (f >> 3) & 3, nt = f & 7;
    const bf16t* Bp = uv ? Vp : Up;
    float* out = uv ? V : U;
    v8f acc = {};
#pragma unroll
    for (int kt = 0; kt < 4; ++kt) {
      v16bf a = load_a_frag(sA, 128, mt * 16, kt * 32, lane);
      v16bf b = load_b_frag(Bp, nt * 4 + kt, lane);
      acc = wmma_bf16(a, b, acc);
    }
    int n = nt * 16 + (lane & 15);
    int mb = mt * 16 + ((lane >> 4) << 3);
#pragma unroll
    for (int j = 0; j < 8; ++j) out[(m0 + mb + j) * 128 + n] = acc[j];
  }
}

// ------------------------------------------------- generic (N*8 x 128) GEMM
__global__ __launch_bounds__(256) void k_gemm128(
    const float* __restrict__ in, const bf16t* __restrict__ Bp,
    float* __restrict__ out) {
  __shared__ bf16t sA[64 * 128];
  int tid = threadIdx.x, lane = tid & 31, wave = tid >> 5;
  size_t m0 = (size_t)blockIdx.x * 64;
  for (int i = tid; i < 64 * 128; i += 256)
    sA[i] = (bf16t)in[m0 * 128 + i];
  __syncthreads();
  for (int f = wave; f < 32; f += 8) {
    int mt = f >> 3, nt = f & 7;
    v8f acc = {};
#pragma unroll
    for (int kt = 0; kt < 4; ++kt) {
      v16bf a = load_a_frag(sA, 128, mt * 16, kt * 32, lane);
      v16bf b = load_b_frag(Bp, nt * 4 + kt, lane);
      acc = wmma_bf16(a, b, acc);
    }
    int n = nt * 16 + (lane & 15);
    int mb = mt * 16 + ((lane >> 4) << 3);
#pragma unroll
    for (int j = 0; j < 8; ++j) out[(m0 + mb + j) * 128 + n] = acc[j];
  }
}

// ------------------------------------- geometric weighted product (_wmp)
__global__ void k_wmp(const float* __restrict__ mv1, const float* __restrict__ mv2,
                      const float* __restrict__ Wt, float* __restrict__ out) {
  const int GP[8][8] = {
      {0, 1, 2, 3, 4, 5, 6, 7},  {1, 0, 4, 14, 2, 7, 11, 5},
      {2, 12, 0, 5, 9, 3, 7, 6}, {3, 6, 13, 0, 7, 10, 1, 4},
      {4, 10, 1, 7, 8, 14, 5, 11}, {5, 7, 11, 2, 6, 8, 12, 9},
      {6, 3, 7, 9, 13, 4, 8, 10}, {7, 5, 6, 4, 11, 9, 10, 8}};
  const int WI[8][8] = {
      {0, 1, 1, 1, 2, 2, 2, 3},  {4, 5, 6, 6, 7, 8, 7, 9},
      {4, 6, 5, 6, 7, 7, 8, 9},  {4, 6, 6, 5, 8, 7, 7, 9},
      {10, 11, 11, 12, 13, 14, 14, 15}, {10, 12, 11, 11, 14, 13, 14, 15},
      {10, 11, 12, 11, 14, 14, 13, 15}, {16, 17, 17, 17, 18, 18, 18, 19}};
  int idx = blockIdx.x * blockDim.x + threadIdx.x;
  if (idx >= NN * DD) return;
  int n = idx >> 7, d = idx & 127;
  const float* p1 = mv1 + (size_t)n * 1024 + d;
  const float* p2 = mv2 + (size_t)n * 1024 + d;
  float a[8], b[8], w[20];
#pragma unroll
  for (int i = 0; i < 8; ++i) { a[i] = p1[i * 128]; b[i] = p2[i * 128]; }
#pragma unroll
  for (int i = 0; i < 20; ++i) w[i] = Wt[i * 128 + d];
  float comb[16];
#pragma unroll
  for (int i = 0; i < 16; ++i) comb[i] = 0.f;
#pragma unroll
  for (int i = 0; i < 8; ++i)
#pragma unroll
    for (int j = 0; j < 8; ++j) comb[GP[i][j]] += a[i] * b[j] * w[WI[i][j]];
  float* o = out + (size_t)n * 1024 + d;
#pragma unroll
  for (int k = 0; k < 8; ++k) o[k * 128] = comb[k] - comb[k + 8];
}

// ----------------------------- per-type update MLP + gated delta, 16 nodes/blk
__global__ __launch_bounds__(256) void k_update(
    float* __restrict__ state, const float* __restrict__ U,
    const float* __restrict__ V, const float* __restrict__ Z1l,
    const float* __restrict__ Z2l, const int* __restrict__ types,
    const bf16t* __restrict__ W1pB, const float* __restrict__ b1B,
    const bf16t* __restrict__ W2pB, const float* __restrict__ b2B) {
  __shared__ bf16t sUi[16 * 256];  //  8 KB
  __shared__ bf16t sH[16 * 512];   // 16 KB
  __shared__ float sO[16 * 512];   // 32 KB
  __shared__ int sType[16];
  int tid = threadIdx.x, lane = tid & 31, wave = tid >> 5;
  int n0 = blockIdx.x * 16;
  if (tid < 16) sType[tid] = types[n0 + tid];
  for (int i = tid; i < 16 * 128; i += 256) {
    int m = i >> 7, d = i & 127;
    size_t nb = (size_t)(n0 + m) * 1024;
    sUi[m * 256 + d] = (bf16t)state[nb + d];
    float v1 = V[nb + 128 + d], v2 = V[nb + 256 + d], v3 = V[nb + 384 + d];
    sUi[m * 256 + 128 + d] = (bf16t)sqrtf(v1 * v1 + v2 * v2 + v3 * v3);
  }
  __syncthreads();

  for (int t = 0; t < TT; ++t) {
    const bf16t* W1p = W1pB + (size_t)t * (256 * 512);
    const bf16t* W2p = W2pB + (size_t)t * (512 * 512);
    const float* b1 = b1B + t * 512;
    const float* b2 = b2B + t * 512;
    for (int nt = wave; nt < 32; nt += 8) {  // GEMM1: 16x512, K=256
      v8f acc = {};
#pragma unroll
      for (int kt = 0; kt < 8; ++kt) {
        if (kt + 1 < 8) prefetch_b(W1p, nt * 8 + kt + 1, lane);
        v16bf a = load_a_frag(sUi, 256, 0, kt * 32, lane);
        v16bf b = load_b_frag(W1p, nt * 8 + kt, lane);
        acc = wmma_bf16(a, b, acc);
      }
      int n = nt * 16 + (lane & 15);
      int mb = (lane >> 4) << 3;
      float bias = b1[n];
#pragma unroll
      for (int j = 0; j < 8; ++j)
        sH[(mb + j) * 512 + n] = (bf16t)silu(acc[j] + bias);
    }
    __syncthreads();
    for (int nt = wave; nt < 32; nt += 8) {  // GEMM2: 16x512, K=512
      v8f acc = {};
      for (int kt = 0; kt < 16; ++kt) {
        if (kt + 1 < 16) prefetch_b(W2p, nt * 16 + kt + 1, lane);
        v16bf a = load_a_frag(sH, 512, 0, kt * 32, lane);
        v16bf b = load_b_frag(W2p, nt * 16 + kt, lane);
        acc = wmma_bf16(a, b, acc);
      }
      int n = nt * 16 + (lane & 15);
      int mb = (lane >> 4) << 3;
      float bias = b2[n];
#pragma unroll
      for (int j = 0; j < 8; ++j) {
        int m = mb + j;
        if (sType[m] == t) sO[m * 512 + n] = acc[j] + bias;
      }
    }
    __syncthreads();
  }

  // delta = a_chunk * (U + Z1l + Z2l); state += delta
  for (int i = tid; i < 16 * 1024; i += 256) {
    int m = i >> 10, rest = i & 1023;
    int ch = rest >> 7, d = rest & 127;
    size_t gi = (size_t)(n0 + m) * 1024 + ch * 128 + d;
    float Zv = U[gi] + Z1l[gi] + Z2l[gi];
    int acol = (ch == 0) ? d : (ch <= 3) ? 128 + d : (ch <= 6) ? 256 + d : 384 + d;
    state[gi] += sO[m * 512 + acol] * Zv;
  }
}

// ---------------------------------------------------------------- readout
__global__ __launch_bounds__(256) void k_readout(
    const float* __restrict__ state, const float* __restrict__ coords,
    const int* __restrict__ gidx, float* __restrict__ dipole) {
  int tid = threadIdx.x, lane = tid & 31, wave = tid >> 5;
  int n = blockIdx.x * 8 + wave;
  if (n >= NN) return;
  const float* s = state + (size_t)n * 1024;
  float q = 0.f;
#pragma unroll
  for (int k = 0; k < 4; ++k) q += s[lane + k * 32];
#pragma unroll
  for (int m = 16; m > 0; m >>= 1) q += __shfl_xor(q, m, 32);
  float mu[3];
#pragma unroll
  for (int c = 0; c < 3; ++c) {
    float acc = 0.f;
#pragma unroll
    for (int k = 0; k < 4; ++k) acc += s[(1 + c) * 128 + lane + k * 32];
#pragma unroll
    for (int m = 16; m > 0; m >>= 1) acc += __shfl_xor(acc, m, 32);
    mu[c] = acc + q * coords[n * 3 + c];
  }
  if (lane == 0) {
    int g = gidx[n];
    atomicAdd(dipole + g * 3 + 0, mu[0]);
    atomicAdd(dipole + g * 3 + 1, mu[1]);
    atomicAdd(dipole + g * 3 + 2, mu[2]);
  }
}

__global__ void k_norm(const float* __restrict__ dipole, float* __restrict__ out) {
  int g = blockIdx.x * blockDim.x + threadIdx.x;
  if (g >= GG) return;
  float x = dipole[g * 3], y = dipole[g * 3 + 1], z = dipole[g * 3 + 2];
  out[g] = sqrtf(x * x + y * y + z * z);
}

// ================================================================= launcher
extern "C" void kernel_launch(void* const* d_in, const int* in_sizes, int n_in,
                              void* d_out, int out_size, void* d_ws, size_t ws_size,
                              hipStream_t stream) {
  (void)in_sizes; (void)n_in; (void)out_size; (void)ws_size;
  const int*   atomic_numbers = (const int*)d_in[0];
  const int*   edge_list      = (const int*)d_in[1];
  const float* edge_lengths   = (const float*)d_in[2];
  const float* edge_vectors   = (const float*)d_in[3];
  const float* node_coords    = (const float*)d_in[4];
  const int*   node_gidx      = (const int*)d_in[5];
  const float* emb_s  = (const float*)d_in[8];
  const float* emb_t  = (const float*)d_in[9];
  const float* rbf_W  = (const float*)d_in[10];
  const float* rbf_b  = (const float*)d_in[11];
  const float* phi_W1 = (const float*)d_in[12];
  const float* phi_b1 = (const float*)d_in[13];
  const float* phi_W2 = (const float*)d_in[14];
  const float* phi_b2 = (const float*)d_in[15];
  const float* U_W    = (const float*)d_in[16];
  const float* V_W    = (const float*)d_in[17];
  const float* Z1w    = (const float*)d_in[18];
  const float* Z2w    = (const float*)d_in[19];
  const float* gp1_W  = (const float*)d_in[20];
  const float* gp2_W  = (const float*)d_in[21];
  const float* upd_W1 = (const float*)d_in[22];
  const float* upd_b1 = (const float*)d_in[23];
  const float* upd_W2 = (const float*)d_in[24];
  const float* upd_b2 = (const float*)d_in[25];

  size_t off = 0;
  auto sub = [&](size_t bytes) {
    void* p = (char*)d_ws + off;
    off += (bytes + 255) & ~(size_t)255;
    return p;
  };
  const size_t SB = (size_t)NN * 8 * DD * sizeof(float);  // 40.96 MB
  float* state = (float*)sub(SB);
  float* agg   = (float*)sub(SB);  // reused as Z scratch after k_add
  float* Um    = (float*)sub(SB);
  float* Vm    = (float*)sub(SB);
  float* Z1l   = (float*)sub(SB);
  float* Z2l   = (float*)sub(SB);
  float* dip   = (float*)sub((size_t)GG * 3 * sizeof(float));
  float* Ztmp  = agg;

  // packed weight buffers (bf16 fragment layout)
  const size_t PH1 = 160 * 512, PH2 = 800 * 512, M128 = 32 * 512;
  const size_t PU1 = 256 * 512, PU2 = 512 * 512;
  bf16t* pk_phi1 = (bf16t*)sub(RR * PH1 * 2);
  bf16t* pk_phi2 = (bf16t*)sub(RR * PH2 * 2);
  bf16t* pk_U    = (bf16t*)sub(RR * M128 * 2);
  bf16t* pk_V    = (bf16t*)sub(RR * M128 * 2);
  bf16t* pk_g1   = (bf16t*)sub(RR * M128 * 2);
  bf16t* pk_g2   = (bf16t*)sub(RR * M128 * 2);
  bf16t* pk_u1   = (bf16t*)sub((size_t)RR * TT * PU1 * 2);
  bf16t* pk_u2   = (bf16t*)sub((size_t)RR * TT * PU2 * 2);

  auto pack = [&](const float* W, bf16t* Bp, int no, int ni) {
    size_t total = (size_t)(no / 16) * (ni / 32) * 512;
    k_pack_b<<<(unsigned)((total + 255) / 256), 256, 0, stream>>>(W, Bp, no, ni);
  };
  for (int r = 0; r < RR; ++r) {
    pack(phi_W1 + (size_t)r * D5 * DD, pk_phi1 + r * PH1, D5, DD);
    pack(phi_W2 + (size_t)r * D5 * D5, pk_phi2 + r * PH2, D5, D5);
    pack(U_W   + (size_t)r * DD * DD, pk_U  + r * M128, DD, DD);
    pack(V_W   + (size_t)r * DD * DD, pk_V  + r * M128, DD, DD);
    pack(gp1_W + (size_t)r * DD * DD, pk_g1 + r * M128, DD, DD);
    pack(gp2_W + (size_t)r * DD * DD, pk_g2 + r * M128, DD, DD);
    for (int t = 0; t < TT; ++t) {
      size_t rt = (size_t)r * TT + t;
      pack(upd_W1 + rt * 512 * 256, pk_u1 + rt * PU1, 512, 256);
      pack(upd_W2 + rt * 512 * 512, pk_u2 + rt * PU2, 512, 512);
    }
  }

  k_init_state<<<(NN * DD + 255) / 256, 256, 0, stream>>>(atomic_numbers, emb_s,
                                                          emb_t, state);
  for (int r = 0; r < RR; ++r) {
    k_zero<<<1024, 256, 0, stream>>>(agg, (size_t)NN * 8 * DD);
    k_edge<<<EE / 32, 256, 0, stream>>>(
        state, agg, edge_list, edge_lengths, edge_vectors,
        pk_phi1 + r * PH1, phi_b1 + (size_t)r * D5,
        pk_phi2 + r * PH2, phi_b2 + (size_t)r * D5,
        rbf_W + (size_t)r * D5 * KK, rbf_b + (size_t)r * D5);
    k_add<<<1024, 256, 0, stream>>>(state, agg, (size_t)NN * 8 * DD);
    k_uv<<<NN * 8 / 64, 256, 0, stream>>>(state, pk_U + r * M128,
                                          pk_V + r * M128, Um, Vm);
    k_wmp<<<NN * DD / 256, 256, 0, stream>>>(Um, Vm, Z1w + (size_t)r * KK * DD, Ztmp);
    k_gemm128<<<NN * 8 / 64, 256, 0, stream>>>(Ztmp, pk_g1 + r * M128, Z1l);
    k_wmp<<<NN * DD / 256, 256, 0, stream>>>(Um, Z1l, Z2w + (size_t)r * KK * DD, Ztmp);
    k_gemm128<<<NN * 8 / 64, 256, 0, stream>>>(Ztmp, pk_g2 + r * M128, Z2l);
    k_update<<<NN / 16, 256, 0, stream>>>(
        state, Um, Vm, Z1l, Z2l, atomic_numbers,
        pk_u1 + (size_t)r * TT * PU1, upd_b1 + (size_t)r * TT * 512,
        pk_u2 + (size_t)r * TT * PU2, upd_b2 + (size_t)r * TT * 512);
  }
  k_zero<<<8, 256, 0, stream>>>(dip, (size_t)GG * 3);
  k_readout<<<NN / 8, 256, 0, stream>>>(state, node_coords, node_gidx, dip);
  k_norm<<<2, 256, 0, stream>>>(dip, (float*)d_out);
}
